// BahdanauAttention_79087527789202
// MI455X (gfx1250) — compile-verified, matrix operand order checked by reading, not run on hardware
//
#include <hip/hip_runtime.h>

// ---------------------------------------------------------------------------
// Bahdanau attention for MI455X (gfx1250, wave32, WMMA).
//   B=4, T=128, S=1024, D=512, U=128
// GEMMs -> v_wmma_f32_16x16x32_bf16 (f32 accum), scores -> hardware v_tanh_f32.
// ---------------------------------------------------------------------------

#define BATCH  4
#define TDEC   128
#define SENC   1024
#define DMODEL 512
#define UNITS  128

typedef __attribute__((ext_vector_type(16))) __bf16 v16bf;
typedef __attribute__((ext_vector_type(8)))  float  v8f;

union BF16Frag { v16bf v; __bf16 e[16]; };
union F32Acc   { v8f   v; float  e[8];  };

__device__ __forceinline__ float fast_tanh(float x) {
#if __has_builtin(__builtin_amdgcn_tanhf)
    return __builtin_amdgcn_tanhf(x);          // gfx1250 v_tanh_f32
#elif __has_builtin(__builtin_amdgcn_tanh_f32)
    return __builtin_amdgcn_tanh_f32(x);
#else
    float y;
    asm("v_tanh_f32 %0, %1" : "=v"(y) : "v"(x));
    return y;
#endif
}

// ---------------------------------------------------------------------------
// Projection GEMM:  C[M,128] = A[M,512] @ W[512,128] + bias
// One 16x16 output tile per wave; 8 waves (256 thr) cover all 128 columns.
// K consumed 32 at a time via v_wmma_f32_16x16x32_bf16.
// ---------------------------------------------------------------------------
__global__ __launch_bounds__(256)
void proj_gemm_bf16_wmma(const float* __restrict__ A,
                         const float* __restrict__ W,
                         const float* __restrict__ bias,
                         float* __restrict__ C, int M) {
    const int lane  = threadIdx.x & 31;
    const int wave  = threadIdx.x >> 5;
    const int m0    = blockIdx.x * 16;
    if (m0 >= M) return;                       // uniform per block
    const int n0    = wave * 16;
    const int half  = lane >> 4;               // 0: lanes 0-15, 1: lanes 16-31
    const int mr    = m0 + (lane & 15);        // A row this lane loads
    const int nc    = n0 + (lane & 15);        // B column this lane loads
    const int kbase = half * 8;                // 16-bit A layout k-offset

    F32Acc acc;
#pragma unroll
    for (int i = 0; i < 8; ++i) acc.e[i] = 0.0f;

    for (int kk = 0; kk < DMODEL; kk += 32) {
        BF16Frag a, b;
#pragma unroll
        for (int p = 0; p < 8; ++p) {
            // ISA 16-bit A layout: lanes0-15 hold K 0-7 (V0-3) and 16-23 (V4-7);
            // lanes16-31 hold K 8-15 and 24-31. Pairs of consecutive K per dword.
            const int k = kk + kbase + 2 * p + ((2 * p) >= 8 ? 8 : 0);
            const float2 av = *(const float2*)(A + (size_t)mr * DMODEL + k);
            a.e[2 * p]     = (__bf16)av.x;
            a.e[2 * p + 1] = (__bf16)av.y;
            // B (32x16): same per-lane K mapping, lane selects column nc.
            b.e[2 * p]     = (__bf16)W[(size_t)k       * UNITS + nc];
            b.e[2 * p + 1] = (__bf16)W[(size_t)(k + 1) * UNITS + nc];
        }
        acc.v = __builtin_amdgcn_wmma_f32_16x16x32_bf16(
            false, a.v, false, b.v, (short)0, acc.v, false, false);
    }

    // C/D layout: VGPR r -> row (half*8 + r), col = lane&15.
#pragma unroll
    for (int r = 0; r < 8; ++r) {
        const int row = m0 + half * 8 + r;
        C[(size_t)row * UNITS + nc] = acc.e[r] + bias[nc];
    }
}

// ---------------------------------------------------------------------------
// Scores + softmax. One block per (b,t). 67M tanh total -> hardware v_tanh_f32.
//   score[s] = v . tanh(w_enc[b,s,:] + w_dec[b,t,:]) + v_b
//   attn[b,t,:] = softmax(score)
// ---------------------------------------------------------------------------
__global__ __launch_bounds__(256)
void scores_softmax(const float* __restrict__ w_enc,
                    const float* __restrict__ w_dec,
                    const float* __restrict__ v,
                    const float* __restrict__ v_b,
                    float* __restrict__ attn) {
    __shared__ float wd_sh[UNITS];
    __shared__ float v_sh[UNITS];
    __shared__ float sc[SENC];
    __shared__ float red[256];

    const int bt  = blockIdx.x;            // 0 .. B*T-1
    const int b   = bt / TDEC;
    const int tid = threadIdx.x;

    if (tid < UNITS) {
        wd_sh[tid] = w_dec[(size_t)bt * UNITS + tid];
        v_sh[tid]  = v[tid];
    }
    __syncthreads();

    const float vb = v_b[0];
    float lmax = -3.0e38f;
#pragma unroll
    for (int j = 0; j < SENC / 256; ++j) {
        const int s = tid + j * 256;
        const float4* we = (const float4*)(w_enc + ((size_t)b * SENC + s) * UNITS);
        float accv = 0.0f;
#pragma unroll 4
        for (int u4 = 0; u4 < UNITS / 4; ++u4) {
            const float4 e  = we[u4];
            const float4 d  = *(const float4*)(wd_sh + u4 * 4);
            const float4 vv = *(const float4*)(v_sh  + u4 * 4);
            accv += vv.x * fast_tanh(e.x + d.x);
            accv += vv.y * fast_tanh(e.y + d.y);
            accv += vv.z * fast_tanh(e.z + d.z);
            accv += vv.w * fast_tanh(e.w + d.w);
        }
        const float scv = accv + vb;
        sc[s] = scv;
        lmax  = fmaxf(lmax, scv);
    }

    // block max
    red[tid] = lmax;
    __syncthreads();
    for (int off = 128; off > 0; off >>= 1) {
        if (tid < off) red[tid] = fmaxf(red[tid], red[tid + off]);
        __syncthreads();
    }
    const float mv = red[0];
    __syncthreads();

    // exp + block sum
    float lsum = 0.0f;
#pragma unroll
    for (int j = 0; j < SENC / 256; ++j) {
        const int s = tid + j * 256;
        const float e = __expf(sc[s] - mv);
        sc[s] = e;
        lsum += e;
    }
    red[tid] = lsum;
    __syncthreads();
    for (int off = 128; off > 0; off >>= 1) {
        if (tid < off) red[tid] += red[tid + off];
        __syncthreads();
    }
    const float inv = 1.0f / red[0];

    float* arow = attn + (size_t)bt * SENC;
#pragma unroll
    for (int j = 0; j < SENC / 256; ++j) {
        const int s = tid + j * 256;
        arow[s] = sc[s] * inv;
    }
}

// ---------------------------------------------------------------------------
// Context GEMM (per batch):  ctx[T,D] = attn[T,S] @ enc[S,D]
// Grid = B * (T/16) * (D/128); 8 waves per block each own a 16x16 tile.
// ---------------------------------------------------------------------------
__global__ __launch_bounds__(256)
void context_gemm_bf16_wmma(const float* __restrict__ attn,
                            const float* __restrict__ enc,
                            float* __restrict__ ctx) {
    const int lane = threadIdx.x & 31;
    const int wave = threadIdx.x >> 5;
    const int bidx = blockIdx.x;               // 4 * 8 * 4 = 128 blocks
    const int b    = bidx >> 5;                // 32 blocks per batch
    const int rem  = bidx & 31;
    const int m0   = (rem >> 2) * 16;          // T tile
    const int n0   = (rem & 3) * 128 + wave * 16;  // D column tile
    const int half  = lane >> 4;
    const int mr    = m0 + (lane & 15);
    const int nc    = n0 + (lane & 15);
    const int kbase = half * 8;

    const float* A  = attn + (size_t)b * TDEC * SENC;    // [128,1024]
    const float* Bm = enc  + (size_t)b * SENC * DMODEL;  // [1024,512]

    F32Acc acc;
#pragma unroll
    for (int i = 0; i < 8; ++i) acc.e[i] = 0.0f;

    for (int kk = 0; kk < SENC; kk += 32) {
        BF16Frag a, bfr;
#pragma unroll
        for (int p = 0; p < 8; ++p) {
            const int k = kk + kbase + 2 * p + ((2 * p) >= 8 ? 8 : 0);
            const float2 av = *(const float2*)(A + (size_t)mr * SENC + k);
            a.e[2 * p]       = (__bf16)av.x;
            a.e[2 * p + 1]   = (__bf16)av.y;
            bfr.e[2 * p]     = (__bf16)Bm[(size_t)k       * DMODEL + nc];
            bfr.e[2 * p + 1] = (__bf16)Bm[(size_t)(k + 1) * DMODEL + nc];
        }
        acc.v = __builtin_amdgcn_wmma_f32_16x16x32_bf16(
            false, a.v, false, bfr.v, (short)0, acc.v, false, false);
    }

#pragma unroll
    for (int r = 0; r < 8; ++r) {
        const int row = m0 + half * 8 + r;
        ctx[((size_t)b * TDEC + row) * DMODEL + nc] = acc.e[r];
    }
}

// ---------------------------------------------------------------------------
extern "C" void kernel_launch(void* const* d_in, const int* in_sizes, int n_in,
                              void* d_out, int out_size, void* d_ws, size_t ws_size,
                              hipStream_t stream) {
    (void)in_sizes; (void)n_in; (void)out_size; (void)ws_size;

    const float* dec = (const float*)d_in[0];   // [4,128,512]
    const float* enc = (const float*)d_in[1];   // [4,1024,512]
    const float* W1  = (const float*)d_in[2];   // [512,128]
    const float* b1  = (const float*)d_in[3];   // [128]
    const float* W2  = (const float*)d_in[4];   // [512,128]
    const float* b2  = (const float*)d_in[5];   // [128]
    const float* v   = (const float*)d_in[6];   // [128]
    const float* vb  = (const float*)d_in[7];   // [1]

    float* ctx_out  = (float*)d_out;                                  // [4,128,512]
    float* attn_out = (float*)d_out + (size_t)BATCH * TDEC * DMODEL;  // [4,128,1024]

    float* w_enc = (float*)d_ws;                            // [4,1024,128] = 2 MB
    float* w_dec = w_enc + (size_t)BATCH * SENC * UNITS;    // [4,128,128]  = 256 KB

    // 1) encoder projection (loop-invariant, hoisted like the reference)
    proj_gemm_bf16_wmma<<<(BATCH * SENC) / 16, 256, 0, stream>>>(
        enc, W1, b1, w_enc, BATCH * SENC);
    // 2) decoder projection
    proj_gemm_bf16_wmma<<<(BATCH * TDEC) / 16, 256, 0, stream>>>(
        dec, W2, b2, w_dec, BATCH * TDEC);
    // 3) additive scores + softmax -> attention weights (second output region)
    scores_softmax<<<BATCH * TDEC, 256, 0, stream>>>(
        w_enc, w_dec, v, vb, attn_out);
    // 4) context = attn @ enc (first output region)
    context_gemm_bf16_wmma<<<BATCH * (TDEC / 16) * (DMODEL / 128), 256, 0, stream>>>(
        attn_out, enc, ctx_out);
}
